// TransformerBlock_14113262534990
// MI455X (gfx1250) — compile-verified, matrix operand order checked by reading, not run on hardware
//
#include <hip/hip_runtime.h>
#include <hip/hip_bf16.h>
#include <math.h>

// ---------------------------------------------------------------------------
// Transformer block for MI455X (gfx1250): bf16 WMMA GEMMs + flash attention.
// ---------------------------------------------------------------------------

typedef __attribute__((ext_vector_type(16))) __bf16 bf16x16;
typedef __attribute__((ext_vector_type(8)))  __bf16 bf16x8;
typedef __attribute__((ext_vector_type(8)))  float  f32x8;

#define WMMA_BF16(A_, B_, C_) \
  __builtin_amdgcn_wmma_f32_16x16x32_bf16(false, (A_), false, (B_), (short)0, (C_), false, false)

__device__ __forceinline__ unsigned short f32_to_bf16(float f) {
  unsigned u = __float_as_uint(f);
  u += 0x7FFFu + ((u >> 16) & 1u);   // round-to-nearest-even
  return (unsigned short)(u >> 16);
}

// Load a 16x32 bf16 A/B fragment (CDNA5 layout) from a row-major [rows][ldk]
// bf16 buffer. Per lane: lo chunk = K[k0+kb .. k0+kb+7], hi chunk = +16,
// where kb = (lane>=16) ? 8 : 0.  Two 16-byte loads -> global_load_b128.
__device__ __forceinline__ bf16x16 frag_ld(const unsigned short* base, int ldk,
                                           int row, int k0, int lane) {
  const __bf16* p = reinterpret_cast<const __bf16*>(base) +
                    (size_t)row * (size_t)ldk + (size_t)(k0 + ((lane >> 4) << 3));
  bf16x8 lo = *reinterpret_cast<const bf16x8*>(p);
  bf16x8 hi = *reinterpret_cast<const bf16x8*>(p + 16);
  return __builtin_shufflevector(lo, hi, 0,1,2,3,4,5,6,7,8,9,10,11,12,13,14,15);
}

// --------------------------- weight convert/transpose -----------------------
// w: [K][N] f32 row-major  ->  wt: [N][K] bf16 row-major
__global__ __launch_bounds__(256) void k_transpose_bf16(const float* __restrict__ w,
                                                        unsigned short* __restrict__ wt,
                                                        int K, int N) {
  size_t tid = (size_t)blockIdx.x * 256 + threadIdx.x;
  if (tid >= (size_t)K * N) return;
  int n = (int)(tid / K);
  int k = (int)(tid % K);
  wt[tid] = f32_to_bf16(w[(size_t)k * N + n]);
}

// ------------------------------- layernorm ----------------------------------
// One block per row of 1024; output bf16 (feeds a bf16 GEMM).
__global__ __launch_bounds__(256) void k_layernorm(const float* __restrict__ x,
                                                   const float* __restrict__ g,
                                                   const float* __restrict__ b,
                                                   unsigned short* __restrict__ out) {
  __shared__ float red[256];
  const int row = blockIdx.x;
  const float4 v = reinterpret_cast<const float4*>(x + (size_t)row * 1024)[threadIdx.x];

  red[threadIdx.x] = v.x + v.y + v.z + v.w;
  __syncthreads();
  for (int off = 128; off > 0; off >>= 1) {
    if (threadIdx.x < off) red[threadIdx.x] += red[threadIdx.x + off];
    __syncthreads();
  }
  const float mu = red[0] * (1.0f / 1024.0f);
  __syncthreads();

  const float dx = v.x - mu, dy = v.y - mu, dz = v.z - mu, dw = v.w - mu;
  red[threadIdx.x] = dx * dx + dy * dy + dz * dz + dw * dw;
  __syncthreads();
  for (int off = 128; off > 0; off >>= 1) {
    if (threadIdx.x < off) red[threadIdx.x] += red[threadIdx.x + off];
    __syncthreads();
  }
  const float inv = rsqrtf(red[0] * (1.0f / 1024.0f) + 1e-5f);

  const float4 gv = reinterpret_cast<const float4*>(g)[threadIdx.x];
  const float4 bv = reinterpret_cast<const float4*>(b)[threadIdx.x];
  unsigned short* op = out + (size_t)row * 1024 + threadIdx.x * 4;
  op[0] = f32_to_bf16(dx * inv * gv.x + bv.x);
  op[1] = f32_to_bf16(dy * inv * gv.y + bv.y);
  op[2] = f32_to_bf16(dz * inv * gv.z + bv.z);
  op[3] = f32_to_bf16(dw * inv * gv.w + bv.w);
}

// ------------------------------ bf16 WMMA GEMM ------------------------------
// C[M,N] = A[M,K](bf16) @ Bt[N,K](bf16, pre-transposed) + bias (+residual)(+GELU)
// Block = 256 threads = 8 waves arranged 2(M)x4(N); each wave owns a 32x32 tile.
// Block tile = 64(M) x 128(N).  All problem dims divide these exactly.
__global__ __launch_bounds__(256) void k_gemm_wmma(const unsigned short* __restrict__ A,
                                                   const unsigned short* __restrict__ Bt,
                                                   const float* __restrict__ bias,
                                                   const float* __restrict__ residual,
                                                   float* __restrict__ outF,
                                                   unsigned short* __restrict__ outH,
                                                   int M, int N, int K, int act) {
  const int lane = threadIdx.x & 31;
  const int wave = threadIdx.x >> 5;
  const int m0 = blockIdx.y * 64 + (wave >> 2) * 32;
  const int n0 = blockIdx.x * 128 + (wave & 3) * 32;
  const int rl = lane & 15;

  f32x8 c00 = {}, c01 = {}, c10 = {}, c11 = {};

  for (int k0 = 0; k0 < K; k0 += 32) {
    // Warm L2/L0 for the next K-slab (gfx1250 global_prefetch_b8; speculative
    // prefetch past the end is silently dropped).
    __builtin_prefetch(reinterpret_cast<const __bf16*>(A) + (size_t)(m0 + rl) * K + k0 + 32, 0, 1);
    __builtin_prefetch(reinterpret_cast<const __bf16*>(Bt) + (size_t)(n0 + rl) * K + k0 + 32, 0, 1);

    bf16x16 a0 = frag_ld(A,  K, m0 + rl,      k0, lane);
    bf16x16 a1 = frag_ld(A,  K, m0 + 16 + rl, k0, lane);
    bf16x16 b0 = frag_ld(Bt, K, n0 + rl,      k0, lane);
    bf16x16 b1 = frag_ld(Bt, K, n0 + 16 + rl, k0, lane);

    c00 = WMMA_BF16(a0, b0, c00);
    c01 = WMMA_BF16(a0, b1, c01);
    c10 = WMMA_BF16(a1, b0, c10);
    c11 = WMMA_BF16(a1, b1, c11);
  }

  // Epilogue. C/D layout: VGPR r holds (row m0+r, col=lane) for lanes 0-15 and
  // (row m0+8+r, col=lane-16) for lanes 16-31.
  const int rhi = (lane >> 4) << 3;
  f32x8* tiles[4] = { &c00, &c01, &c10, &c11 };
#pragma unroll
  for (int t = 0; t < 4; ++t) {
    const int mb = m0 + (t >> 1) * 16;
    const int nb = n0 + (t & 1) * 16;
    const int col = nb + rl;
    const float bcol = bias[col];
#pragma unroll
    for (int r = 0; r < 8; ++r) {
      const int row = mb + r + rhi;
      float v = (*tiles[t])[r] + bcol;
      if (residual) v += residual[(size_t)row * N + col];
      if (act) v = 0.5f * v * (1.0f + erff(v * 0.70710678118654752f));  // exact GELU
      if (outF) outF[(size_t)row * N + col] = v;
      else      outH[(size_t)row * N + col] = f32_to_bf16(v);
    }
  }
}

// ------------------------- qkv split / v-transpose --------------------------
// qkv: [4096][3072] bf16  ->  q,k: [B*H][2048][64] bf16,  vt: [B*H][64][2048]
__global__ __launch_bounds__(256) void k_split_qkv(const unsigned short* __restrict__ qkv,
                                                   unsigned short* __restrict__ q,
                                                   unsigned short* __restrict__ k,
                                                   unsigned short* __restrict__ vt) {
  size_t tid = (size_t)blockIdx.x * 256 + threadIdx.x;
  if (tid >= (size_t)4096 * 3072) return;
  const int row = (int)(tid / 3072);       // global token 0..4095
  const int c   = (int)(tid % 3072);
  const int which = c >> 10;               // 0=q 1=k 2=v
  const int cc = c & 1023;
  const int h = cc >> 6, d = cc & 63;
  const int b = row >> 11, n = row & 2047;
  const int bh = b * 16 + h;
  const unsigned short v = qkv[tid];
  if (which == 0)      q [((size_t)bh * 2048 + n) * 64 + d]  = v;
  else if (which == 1) k [((size_t)bh * 2048 + n) * 64 + d]  = v;
  else                 vt[((size_t)bh * 64 + d) * 2048 + n]  = v;
}

// --------------------------- flash attention --------------------------------
// One wave per 16-query tile (8 waves/block). Online softmax; S = Q@K^T via
// WMMA (reduction over D=64, two k-steps), P re-shaped through a per-wave LDS
// slice (ds_load_b128 back as A-fragment), O += P@V via WMMA over V^T.
__global__ __launch_bounds__(256) void k_flash_attn(const unsigned short* __restrict__ Q,
                                                    const unsigned short* __restrict__ Kb,
                                                    const unsigned short* __restrict__ Vt,
                                                    unsigned short* __restrict__ O) {
  const int lane = threadIdx.x & 31;
  const int wave = threadIdx.x >> 5;
  const int qt = blockIdx.x * 8 + wave;      // 0..4095 query tiles
  const int bh = qt >> 7;                    // batch*head 0..31
  const int qrow0 = (qt & 127) << 4;
  const int rl = lane & 15;
  const int rhi = (lane >> 4) << 3;

  const unsigned short* qp = Q  + ((size_t)bh * 2048 + qrow0) * 64;
  const unsigned short* kp = Kb + (size_t)bh * 2048 * 64;
  const unsigned short* vp = Vt + (size_t)bh * 64 * 2048;

  __shared__ alignas(16) unsigned short Pbuf[8][16 * 32];
  unsigned short* pb = &Pbuf[wave][0];

  // Q fragments are invariant over the key loop.
  const bf16x16 aq0 = frag_ld(qp, 64, rl, 0,  lane);
  const bf16x16 aq1 = frag_ld(qp, 64, rl, 32, lane);

  float mrow[8], lrow[8];
  f32x8 acc0 = {}, acc1 = {}, acc2 = {}, acc3 = {};
#pragma unroll
  for (int r = 0; r < 8; ++r) { mrow[r] = -1e30f; lrow[r] = 0.0f; }

  for (int kt = 0; kt < 2048; kt += 32) {
    // S tiles: 16 queries x 32 keys as two 16x16 WMMAs, reduction over d=64.
    bf16x16 bk00 = frag_ld(kp, 64, kt + rl,      0,  lane);
    bf16x16 bk01 = frag_ld(kp, 64, kt + rl,      32, lane);
    bf16x16 bk10 = frag_ld(kp, 64, kt + 16 + rl, 0,  lane);
    bf16x16 bk11 = frag_ld(kp, 64, kt + 16 + rl, 32, lane);
    f32x8 s0 = {}, s1 = {};
    s0 = WMMA_BF16(aq0, bk00, s0);
    s0 = WMMA_BF16(aq1, bk01, s0);
    s1 = WMMA_BF16(aq0, bk10, s1);
    s1 = WMMA_BF16(aq1, bk11, s1);

    // Online softmax: per-row max/sum across the 16-lane half-wave.
    float fac[8];
#pragma unroll
    for (int r = 0; r < 8; ++r) {
      float v0 = s0[r] * 0.125f;          // scale = HEAD_DIM^-0.5 = 1/8
      float v1 = s1[r] * 0.125f;
      float mx = fmaxf(v0, v1);
#pragma unroll
      for (int off = 8; off >= 1; off >>= 1) mx = fmaxf(mx, __shfl_xor(mx, off, 16));
      const float mnew = fmaxf(mrow[r], mx);
      const float p0 = __expf(v0 - mnew);
      const float p1 = __expf(v1 - mnew);
      float ps = p0 + p1;
#pragma unroll
      for (int off = 8; off >= 1; off >>= 1) ps += __shfl_xor(ps, off, 16);
      fac[r]  = __expf(mrow[r] - mnew);
      lrow[r] = lrow[r] * fac[r] + ps;
      mrow[r] = mnew;
      const int rr = r + rhi;             // this lane's row within the 16-tile
      pb[rr * 32 + rl]      = f32_to_bf16(p0);
      pb[rr * 32 + 16 + rl] = f32_to_bf16(p1);
    }
#pragma unroll
    for (int r = 0; r < 8; ++r) {
      acc0[r] *= fac[r]; acc1[r] *= fac[r]; acc2[r] *= fac[r]; acc3[r] *= fac[r];
    }

    // Re-read P as a 16x32 A-fragment from this wave's private LDS slice
    // (in-order LDS within a wave; compiler inserts s_wait_dscnt).
    const bf16x16 ap = frag_ld(pb, 32, rl, 0, lane);

    // V^T B-fragments: col = d, reduction along keys (contiguous in vt rows).
    const bf16x16 bv0 = frag_ld(vp, 2048,  0 + rl, kt, lane);
    const bf16x16 bv1 = frag_ld(vp, 2048, 16 + rl, kt, lane);
    const bf16x16 bv2 = frag_ld(vp, 2048, 32 + rl, kt, lane);
    const bf16x16 bv3 = frag_ld(vp, 2048, 48 + rl, kt, lane);
    acc0 = WMMA_BF16(ap, bv0, acc0);
    acc1 = WMMA_BF16(ap, bv1, acc1);
    acc2 = WMMA_BF16(ap, bv2, acc2);
    acc3 = WMMA_BF16(ap, bv3, acc3);
  }

  // Normalize and scatter to o[b][n][h*64+d] (bf16, feeds proj GEMM).
  const int b = bh >> 4, h = bh & 15;
#pragma unroll
  for (int r = 0; r < 8; ++r) {
    const int row = qrow0 + r + rhi;
    const float inv = 1.0f / lrow[r];
    unsigned short* op = O + ((size_t)(b * 2048 + row)) * 1024 + h * 64 + rl;
    op[0]  = f32_to_bf16(acc0[r] * inv);
    op[16] = f32_to_bf16(acc1[r] * inv);
    op[32] = f32_to_bf16(acc2[r] * inv);
    op[48] = f32_to_bf16(acc3[r] * inv);
  }
}

// ------------------------------- launch --------------------------------------
extern "C" void kernel_launch(void* const* d_in, const int* in_sizes, int n_in,
                              void* d_out, int out_size, void* d_ws, size_t ws_size,
                              hipStream_t stream) {
  const float* x      = (const float*)d_in[0];
  const float* ln1_g  = (const float*)d_in[1];
  const float* ln1_b  = (const float*)d_in[2];
  const float* ln2_g  = (const float*)d_in[3];
  const float* ln2_b  = (const float*)d_in[4];
  const float* w_qkv  = (const float*)d_in[5];
  const float* b_qkv  = (const float*)d_in[6];
  const float* w_proj = (const float*)d_in[7];
  const float* b_proj = (const float*)d_in[8];
  const float* w_fc1  = (const float*)d_in[9];
  const float* b_fc1  = (const float*)d_in[10];
  const float* w_fc2  = (const float*)d_in[11];
  const float* b_fc2  = (const float*)d_in[12];
  float* out = (float*)d_out;

  const int M = 4096;  // B*N tokens

  // Workspace carve-up (bytes).
  char* ws = (char*)d_ws;
  size_t off = 0;
  auto carve = [&](size_t bytes) { char* p = ws + off; off += (bytes + 255) & ~(size_t)255; return p; };
  unsigned short* wqkvT = (unsigned short*)carve((size_t)3072 * 1024 * 2);
  unsigned short* wprojT= (unsigned short*)carve((size_t)1024 * 1024 * 2);
  unsigned short* wfc1T = (unsigned short*)carve((size_t)4096 * 1024 * 2);
  unsigned short* wfc2T = (unsigned short*)carve((size_t)1024 * 4096 * 2);
  unsigned short* h1    = (unsigned short*)carve((size_t)M * 1024 * 2);
  unsigned short* qkv   = (unsigned short*)carve((size_t)M * 3072 * 2);
  unsigned short* qb    = (unsigned short*)carve((size_t)M * 1024 * 2);
  unsigned short* kb    = (unsigned short*)carve((size_t)M * 1024 * 2);
  unsigned short* vtb   = (unsigned short*)carve((size_t)M * 1024 * 2);
  unsigned short* ob    = (unsigned short*)carve((size_t)M * 1024 * 2);
  float*          x1    = (float*)         carve((size_t)M * 1024 * 4);
  unsigned short* h2    = (unsigned short*)carve((size_t)M * 1024 * 2);
  unsigned short* mlp   = (unsigned short*)carve((size_t)M * 4096 * 2);
  (void)ws_size; (void)n_in; (void)in_sizes; (void)out_size;

  // 1) Weights -> bf16, transposed to [N][K] for contiguous B-fragments.
  k_transpose_bf16<<<(3072 * 1024 + 255) / 256, 256, 0, stream>>>(w_qkv,  wqkvT, 1024, 3072);
  k_transpose_bf16<<<(1024 * 1024 + 255) / 256, 256, 0, stream>>>(w_proj, wprojT, 1024, 1024);
  k_transpose_bf16<<<(4096 * 1024 + 255) / 256, 256, 0, stream>>>(w_fc1,  wfc1T, 1024, 4096);
  k_transpose_bf16<<<(4096 * 1024 + 255) / 256, 256, 0, stream>>>(w_fc2,  wfc2T, 4096, 1024);

  // 2) LN1 -> bf16
  k_layernorm<<<M, 256, 0, stream>>>(x, ln1_g, ln1_b, h1);

  // 3) qkv = h1 @ w_qkv + b_qkv   [4096 x 3072]
  k_gemm_wmma<<<dim3(3072 / 128, M / 64), 256, 0, stream>>>(
      h1, wqkvT, b_qkv, nullptr, nullptr, qkv, M, 3072, 1024, 0);

  // 4) split into q,k (row-major per head) and v^T (per head)
  k_split_qkv<<<(int)(((size_t)M * 3072 + 255) / 256), 256, 0, stream>>>(qkv, qb, kb, vtb);

  // 5) flash attention -> o [B,N,C] bf16
  k_flash_attn<<<512, 256, 0, stream>>>(qb, kb, vtb, ob);

  // 6) x1 = x + o @ w_proj + b_proj   (fp32)
  k_gemm_wmma<<<dim3(1024 / 128, M / 64), 256, 0, stream>>>(
      ob, wprojT, b_proj, x, x1, nullptr, M, 1024, 1024, 0);

  // 7) LN2 -> bf16
  k_layernorm<<<M, 256, 0, stream>>>(x1, ln2_g, ln2_b, h2);

  // 8) mlp = gelu(h2 @ w_fc1 + b_fc1)   [4096 x 4096] bf16
  k_gemm_wmma<<<dim3(4096 / 128, M / 64), 256, 0, stream>>>(
      h2, wfc1T, b_fc1, nullptr, nullptr, mlp, M, 4096, 1024, 1);

  // 9) out = x1 + mlp @ w_fc2 + b_fc2   (fp32, K=4096)
  k_gemm_wmma<<<dim3(1024 / 128, M / 64), 256, 0, stream>>>(
      mlp, wfc2T, b_fc2, x1, out, nullptr, M, 1024, 4096, 0);
}
